// TernaryLinear_23691039605071
// MI455X (gfx1250) — matrix operand order repaired and checked by reading, not compiled, not run on hardware
//
#include <hip/hip_runtime.h>

typedef __bf16 bf16_t;
typedef __attribute__((ext_vector_type(4)))  bf16_t v4bf;
typedef __attribute__((ext_vector_type(8)))  bf16_t v8bf;
typedef __attribute__((ext_vector_type(16))) bf16_t v16bf;
typedef __attribute__((ext_vector_type(4)))  float  v4f;
typedef __attribute__((ext_vector_type(8)))  float  v8f;

static __device__ __forceinline__ v16bf cat16(v8bf lo, v8bf hi) {
  return __builtin_shufflevector(lo, hi, 0,1,2,3,4,5,6,7,8,9,10,11,12,13,14,15);
}

// ---------------- x: f32 -> bf16 (vectorized x4) ----------------
__global__ void __launch_bounds__(256)
ternlin_cvt_x(const float* __restrict__ x, bf16_t* __restrict__ xb, long n4) {
  long i = (long)blockIdx.x * blockDim.x + threadIdx.x;
  if (i >= n4) return;
  v4f v = ((const v4f*)x)[i];
  v4bf r;
#pragma unroll
  for (int e = 0; e < 4; ++e) r[e] = (bf16_t)v[e];
  ((v4bf*)xb)[i] = r;
}

// ------- w: ternary dequant f32 -> bf16 (branch order = reference) -------
__global__ void __launch_bounds__(256)
ternlin_dequant_w(const float* __restrict__ w, bf16_t* __restrict__ wb, long n4) {
  long i = (long)blockIdx.x * blockDim.x + threadIdx.x;
  if (i >= n4) return;
  v4f v = ((const v4f*)w)[i];
  v4bf r;
#pragma unroll
  for (int e = 0; e < 4; ++e) {
    float f = v[e];
    float d;
    if (f > 0.5f)                       d = 1.0f;
    else if (f < -0.5f)                 d = -1.0f;
    else if (__builtin_fabsf(f) < 0.1f) d = 0.0f;
    else                                d = f + 0.5f;  // (f + 0.5) / (2*0.5)
    r[e] = (bf16_t)d;
  }
  ((v4bf*)wb)[i] = r;
}

// ---------------- WMMA bf16 GEMM: C = A * B^T + bias ----------------
// Block: 256 threads = 8 waves (2 M x 4 N); block tile 128x128; wave tile 64x32.
// K pipelined in BK=64 stages, double-buffered in LDS, filled with
// global_load_async_to_lds_b128 (ASYNCcnt), consumed via ds_load_b128 into
// v_wmma_f32_16x16x32_bf16 fragments.
#define TM 4
#define TN 2
#define BK 64
#define LDS_STRIDE 72   // 64 + 8 bf16 pad: 144B rows -> conflict-free 16B frag reads

__device__ __forceinline__ void async_copy_b128(const bf16_t* g, const bf16_t* l) {
  const unsigned ldsAddr = (unsigned)(unsigned long long)(const void*)l;  // addr[31:0] = LDS offset
  asm volatile("global_load_async_to_lds_b128 %0, %1, off"
               :: "v"(ldsAddr), "v"((unsigned long long)(const void*)g)
               : "memory");
}

// Each thread issues 4 A-chunks + 4 B-chunks (16B each): 256 threads fill
// 128x64 bf16 A and B tiles. Lane-consecutive chunks are K-contiguous.
__device__ __forceinline__ void load_stage(const bf16_t* __restrict__ A,
                                           const bf16_t* __restrict__ B,
                                           bf16_t* sA, bf16_t* sB,
                                           int K, int rowA0, int rowB0,
                                           int k0, int t) {
  const int r = t >> 3;      // 0..31
  const int q = t & 7;       // 16B chunk within a 128B K-row
#pragma unroll
  for (int i = 0; i < 4; ++i) {
    const int row = r + 32 * i;
    async_copy_b128(A + (long)(rowA0 + row) * K + k0 + q * 8,
                    sA + row * LDS_STRIDE + q * 8);
  }
#pragma unroll
  for (int i = 0; i < 4; ++i) {
    const int row = r + 32 * i;
    async_copy_b128(B + (long)(rowB0 + row) * K + k0 + q * 8,
                    sB + row * LDS_STRIDE + q * 8);
  }
}

__global__ void __launch_bounds__(256, 2)
ternlin_gemm_wmma(const bf16_t* __restrict__ A, const bf16_t* __restrict__ B,
                  const float* __restrict__ bias, float* __restrict__ C,
                  int M, int N, int K) {
  __shared__ bf16_t smA[2][128 * LDS_STRIDE];
  __shared__ bf16_t smB[2][128 * LDS_STRIDE];

  const int t      = threadIdx.x;
  const int lane   = t & 31;
  const int wave   = t >> 5;
  const int lane16 = lane & 15;
  const int laneHi = lane >> 4;          // 0: lanes 0-15, 1: lanes 16-31

  const int waveM = wave >> 2;           // 0..1
  const int waveN = wave & 3;            // 0..3

  const int rowA0 = blockIdx.y * 128;    // block M base
  const int rowB0 = blockIdx.x * 128;    // block N base

  v8f acc[TM][TN] = {};

  const int nk = K / BK;
  load_stage(A, B, smA[0], smB[0], K, rowA0, rowB0, 0, t);

  for (int kt = 0; kt < nk; ++kt) {
    if (kt + 1 < nk) {
      load_stage(A, B, smA[(kt + 1) & 1], smB[(kt + 1) & 1],
                 K, rowA0, rowB0, (kt + 1) * BK, t);
      // 8 newer loads outstanding; async loads complete in order, so
      // waiting to <=8 drains the current stage while next stays in flight.
      asm volatile("s_wait_asynccnt 0x8" ::: "memory");
    } else {
      asm volatile("s_wait_asynccnt 0x0" ::: "memory");
    }
    __syncthreads();

    const bf16_t* cA = smA[kt & 1];
    const bf16_t* cB = smB[kt & 1];

#pragma unroll
    for (int ks = 0; ks < BK; ks += 32) {
      // A fragment: row M = waveM*64 + tile*16 + lane16.
      //   lanes 0-15 : K chunks {ks+0..7, ks+16..23}; lanes 16-31: {+8, +24}
      v16bf afrag[TM];
#pragma unroll
      for (int i = 0; i < TM; ++i) {
        const bf16_t* p = cA + (waveM * 64 + i * 16 + lane16) * LDS_STRIDE
                             + ks + (laneHi << 3);
        v8bf lo = *(const v8bf*)(p);
        v8bf hi = *(const v8bf*)(p + 16);
        afrag[i] = cat16(lo, hi);
      }
      // B fragment: col N = waveN*32 + tile*16 + lane16, K chunks {ks+8h..+7, ks+16+8h..+7}.
      v16bf bfrag[TN];
#pragma unroll
      for (int j = 0; j < TN; ++j) {
        const bf16_t* p = cB + (waveN * 32 + j * 16 + lane16) * LDS_STRIDE
                             + ks + (laneHi << 3);
        v8bf lo = *(const v8bf*)(p);
        v8bf hi = *(const v8bf*)(p + 16);
        bfrag[j] = cat16(lo, hi);
      }

#pragma unroll
      for (int i = 0; i < TM; ++i)
#pragma unroll
        for (int j = 0; j < TN; ++j)
          acc[i][j] = __builtin_amdgcn_wmma_f32_16x16x32_bf16(
              false, afrag[i], false, bfrag[j],
              (short)0, acc[i][j], false, false);
    }
    __syncthreads();  // all waves done reading this buffer before it refills
  }

  // C/D layout: VGPR r, lanes 0-15 -> M = r, lanes 16-31 -> M = r + 8; N = lane16.
  const int m0 = rowA0 + waveM * 64;
  const int n0 = rowB0 + waveN * 32;
#pragma unroll
  for (int j = 0; j < TN; ++j) {
    const int col = n0 + j * 16 + lane16;
    const float bs = bias[col];
#pragma unroll
    for (int i = 0; i < TM; ++i) {
      const int rowBase = m0 + i * 16 + laneHi * 8;
#pragma unroll
      for (int r = 0; r < 8; ++r) {
        C[(long)(rowBase + r) * N + col] = acc[i][j][r] + bs;
      }
    }
  }
}

extern "C" void kernel_launch(void* const* d_in, const int* in_sizes, int n_in,
                              void* d_out, int out_size, void* d_ws, size_t ws_size,
                              hipStream_t stream) {
  const float* x    = (const float*)d_in[0];   // [M,K] f32
  const float* wf   = (const float*)d_in[1];   // [N,K] f32
  const float* bias = (const float*)d_in[2];   // [N]   f32
  float* out = (float*)d_out;                  // [M,N] f32

  const int N = in_sizes[2];
  const int K = in_sizes[1] / N;
  const int M = in_sizes[0] / K;

  bf16_t* xb = (bf16_t*)d_ws;                  // M*K bf16
  bf16_t* wb = xb + (size_t)M * K;             // N*K bf16

  {
    long n4 = (long)M * K / 4;
    ternlin_cvt_x<<<(unsigned)((n4 + 255) / 256), 256, 0, stream>>>(x, xb, n4);
  }
  {
    long n4 = (long)N * K / 4;
    ternlin_dequant_w<<<(unsigned)((n4 + 255) / 256), 256, 0, stream>>>(wf, wb, n4);
  }

  dim3 grid((unsigned)(N / 128), (unsigned)(M / 128));
  ternlin_gemm_wmma<<<grid, 256, 0, stream>>>(xb, wb, bias, out, M, N, K);
}
